// BinaryConv2d_30983894073536
// MI455X (gfx1250) — matrix operand order repaired and checked by reading, not compiled, not run on hardware
//
#include <hip/hip_runtime.h>

typedef __attribute__((ext_vector_type(8)))  unsigned short ushort8;
typedef __attribute__((ext_vector_type(16))) unsigned short ushort16;
typedef __attribute__((ext_vector_type(16))) __bf16         v16bf;
typedef __attribute__((ext_vector_type(8)))  float          v8f;

__device__ __forceinline__ v16bf as_bf16x16(ushort16 u) {
    union { ushort16 u; v16bf b; } cvt;
    cvt.u = u;
    return cvt.b;
}

// Round-to-nearest-even fp32 -> bf16 bits (inputs are finite gaussians; no NaN path needed)
__device__ __forceinline__ unsigned short f32_to_bf16_rne(float f) {
    unsigned int u = __float_as_uint(f);
    u += 0x7FFFu + ((u >> 16) & 1u);
    return (unsigned short)(u >> 16);
}
__device__ __forceinline__ float bf16_bits_to_f32(unsigned short h) {
    return __uint_as_float(((unsigned int)h) << 16);
}

// ---------------------------------------------------------------------------
// Kernel 0: binarize W (3,3,64,128) -> bf16 {-1,0,+1}, packed in B-fragment
// order: Wp[((s*2 + g)*128 + n)*16 + t] = sign(W[k, n]) with k = s*32+g*16+t.
// (B 32x16, contiguous-K order used here: lanes 0-15 = N, K 0-15; lanes 16-31 = N, K 16-31; each lane
//  then reads 16 contiguous halves = 32 bytes.)
// ---------------------------------------------------------------------------
__global__ void binarize_pack_w(const float* __restrict__ W,
                                unsigned short* __restrict__ Wp) {
    int tid = blockIdx.x * 256 + threadIdx.x;
    if (tid >= 18 * 2 * 128 * 16) return;          // 73728
    int t  = tid & 15;
    int n  = (tid >> 4) & 127;
    int sg = tid >> 11;                            // 0..35
    int g  = sg & 1;
    int s  = sg >> 1;
    int k  = s * 32 + g * 16 + t;                  // 0..575
    float w = W[(size_t)k * 128 + n];
    unsigned short v = (w > 0.0f) ? 0x3F80u : ((w < 0.0f) ? 0xBF80u : 0u);
    Wp[tid] = v;
}

// ---------------------------------------------------------------------------
// Main kernel: block = 8 waves = 256 threads.
// Block tile: 16 output pixels (one row, 16 consecutive w) x 128 channels.
// Wave w owns N-columns [16w, 16w+16).  K = 576 in 18 steps of 32 (one 3x3
// tap half per step).  A is staged in LDS as bf16 hi/lo planes of the
// 3 x 18 x 64 input halo.
// ---------------------------------------------------------------------------
__global__ __launch_bounds__(256, 1)
void bconv_wmma(const float* __restrict__ x,
                const unsigned short* __restrict__ Wp,
                const float* __restrict__ bias,
                float* __restrict__ out) {
    // [plane][ky][col][c] : 2*3*18*64 ushorts = 13824 B
    __shared__ unsigned short sA[2][3][18][64];

    const int wtile = blockIdx.x;      // 0..13
    const int h     = blockIdx.y;      // 0..223
    const int img   = blockIdx.z;      // 0..15
    const int w0    = wtile * 16;

    // -------- stage input halo, split fp32 -> bf16 hi + lo ----------------
    for (int i = threadIdx.x; i < 3 * 18 * 64; i += 256) {
        int c   = i & 63;
        int col = (i >> 6) % 18;
        int ky  = i / (18 * 64);
        int hh  = h + ky - 1;
        int ww  = w0 + col - 1;
        float v = 0.0f;
        if ((unsigned)hh < 224u && (unsigned)ww < 224u)
            v = x[((((size_t)img * 224 + hh) * 224) + ww) * 64 + c];
        unsigned short hi = f32_to_bf16_rne(v);
        float rem = v - bf16_bits_to_f32(hi);
        unsigned short lo = f32_to_bf16_rne(rem);
        sA[0][ky][col][c] = hi;
        sA[1][ky][col][c] = lo;
    }
    __syncthreads();

    const int lane = threadIdx.x & 31;
    const int wave = threadIdx.x >> 5;         // 0..7
    const int m    = lane & 15;                // output pixel within tile / N col
    const int g    = lane >> 4;                // lane half-group
    const int n    = wave * 16 + m;            // global output channel

    v8f acc = {};

    // K loop: 18 steps of 32; each step covers one tap (ky,kx), c in [cb,cb+32)
#pragma unroll
    for (int s = 0; s < 18; ++s) {
        const int tap = s >> 1;
        const int cb  = (s & 1) * 32;
        const int ky  = tap / 3;
        const int kx  = tap - ky * 3;

        // B fragment: 32 contiguous bytes per lane (L2-resident weights)
        const ushort8* bp =
            (const ushort8*)(Wp + ((((size_t)(s * 2 + g)) * 128 + n) << 4));
        ushort8 b0 = bp[0];
        ushort8 b1 = bp[1];
        v16bf B = as_bf16x16(__builtin_shufflevector(
            b0, b1, 0, 1, 2, 3, 4, 5, 6, 7, 8, 9, 10, 11, 12, 13, 14, 15));

        // A fragments (hi & lo planes): contiguous-K 16x32 order chosen here (not the ISA table order) ->
        // lane (m, g) reads halves at offsets [16g .. 16g+15] (contiguous K, matches B)
        const unsigned short* a0 = &sA[0][ky][m + kx][cb];
        const unsigned short* a1 = &sA[1][ky][m + kx][cb];
        ushort8 h0 = *(const ushort8*)(a0 + 16 * g);
        ushort8 h1 = *(const ushort8*)(a0 + 16 * g + 8);
        ushort8 l0 = *(const ushort8*)(a1 + 16 * g);
        ushort8 l1 = *(const ushort8*)(a1 + 16 * g + 8);
        v16bf Ahi = as_bf16x16(__builtin_shufflevector(
            h0, h1, 0, 1, 2, 3, 4, 5, 6, 7, 8, 9, 10, 11, 12, 13, 14, 15));
        v16bf Alo = as_bf16x16(__builtin_shufflevector(
            l0, l1, 0, 1, 2, 3, 4, 5, 6, 7, 8, 9, 10, 11, 12, 13, 14, 15));

        // D = A_hi*B + (A_lo*B + C): exact products (B in {-1,0,+1}), f32 accum
        acc = __builtin_amdgcn_wmma_f32_16x16x32_bf16(
            false, Alo, false, B, (short)0, acc, false, false);
        acc = __builtin_amdgcn_wmma_f32_16x16x32_bf16(
            false, Ahi, false, B, (short)0, acc, false, false);
    }

    // C/D layout: VGPR j, lane group g -> row M = j + 8g, col N = n
    const float bb = bias[n];
#pragma unroll
    for (int j = 0; j < 8; ++j) {
        int M = j + 8 * g;
        size_t o = ((((size_t)img * 224 + h) * 224) + (w0 + M)) * 128 + n;
        out[o] = acc[j] + bb;
    }
}

// ---------------------------------------------------------------------------
extern "C" void kernel_launch(void* const* d_in, const int* in_sizes, int n_in,
                              void* d_out, int out_size, void* d_ws, size_t ws_size,
                              hipStream_t stream) {
    const float* x = (const float*)d_in[0];   // (16,224,224,64) f32
    const float* W = (const float*)d_in[1];   // (3,3,64,128)    f32
    const float* b = (const float*)d_in[2];   // (128,)          f32
    float* out     = (float*)d_out;           // (16,224,224,128) f32

    unsigned short* Wp = (unsigned short*)d_ws;   // 73728 bf16 = 144 KB

    // Pack binarized weights into WMMA B-fragment order
    binarize_pack_w<<<(18 * 2 * 128 * 16 + 255) / 256, 256, 0, stream>>>(W, Wp);

    // 14 w-tiles x 224 rows x 16 images
    dim3 grid(14, 224, 16);
    bconv_wmma<<<grid, 256, 0, stream>>>(x, Wp, b, out);
}
